// UncertaintyHead_59450937311576
// MI455X (gfx1250) — matrix operand order repaired and checked by reading, not compiled, not run on hardware
//
#include <hip/hip_runtime.h>
#include <hip/hip_bf16.h>
#include <stdint.h>

typedef __attribute__((ext_vector_type(16))) _Float16 v16h;
typedef __attribute__((ext_vector_type(8)))  _Float16 h8;
typedef __attribute__((ext_vector_type(8)))  float    v8f;
typedef __attribute__((ext_vector_type(4)))  float    f32x4;

#define BDIM   32768
#define DIN    512
#define H1     128
#define H2     64
#define MCS    10

static constexpr float    INV_KEEP     = 1.0f / 0.9f;
static constexpr double   KEEP_EFF_D   = 0.3486784401;           // 0.9^10
static constexpr float    INV_KEEP_EFF = 2.8679719907924413f;    // 1/0.9^10
static constexpr uint32_t TH_KEEP      = (uint32_t)(0.9 * 4294967296.0);
static constexpr uint32_t TH_EFF8      = (uint32_t)(KEEP_EFF_D * 256.0); // 89
static constexpr uint32_t SEED_S       = 0x85ebca6bu;
static constexpr uint32_t SEED_MC      = 0x9e3779b9u;

__device__ __forceinline__ uint32_t hash_u32(uint32_t x) {
    x ^= x >> 17; x *= 0xed5ad4bbu;
    x ^= x >> 11; x *= 0xac4c1b51u;
    x ^= x >> 15; x *= 0x31848babu;
    x ^= x >> 14;
    return x;
}

__device__ __forceinline__ v8f wmma_f16(v16h a, v16h b, v8f c) {
    // D(f32 16x16) = A(f16 16x32) * B(f16 32x16) + C
    return __builtin_amdgcn_wmma_f32_16x16x32_f16(
        /*neg_a=*/false, a, /*neg_b=*/false, b,
        /*c_mod=*/(short)0, c, /*reuse_a=*/false, /*reuse_b=*/false);
}

// dst[n*K + k] = (f16) src[k*N + n]   -- transpose + downconvert weights
__global__ void uh_xpose_f16(const float* __restrict__ src,
                             _Float16* __restrict__ dst, int K, int N) {
    int i = blockIdx.x * blockDim.x + threadIdx.x;
    if (i >= K * N) return;
    int k = i / N, n = i - k * N;
    dst[(size_t)n * K + k] = (_Float16)src[i];
}

__global__ __launch_bounds__(256)
void uh_fused(const float* __restrict__ x,
              const float* __restrict__ b1,
              const float* __restrict__ bam,
              const float* __restrict__ bav,
              const float* __restrict__ be,
              const _Float16* __restrict__ w1t,   // [H1][DIN]  = W1^T  f16
              const _Float16* __restrict__ wamt,  // [H2][H1]   = Wam^T f16
              const _Float16* __restrict__ wavt,  // [H2][H1]   = Wav^T f16
              const _Float16* __restrict__ wet,   // [H2][H1]   = We^T  f16
              float* __restrict__ out) {
    __shared__ _Float16 hTile[8][16][H1];   // per-wave 16x128 h tile (32 KB)

    const int lane = threadIdx.x & 31;
    const int wid  = threadIdx.x >> 5;
    const int lh   = lane & 15;             // lane within half-wave
    const int hsel = lane >> 4;             // half select (K-group for A/B)
    const int tile = blockIdx.x * 8 + wid;  // one 16-row tile per wave
    const int row0 = tile * 16;

    float* out_am = out;
    float* out_av = out + 1 * (size_t)BDIM * H2;
    float* out_em = out + 2 * (size_t)BDIM * H2;
    float* out_ev = out + 3 * (size_t)BDIM * H2;

    // ---------------- GEMM1: acc = x @ W1  (16 rows x 128 cols) ------------
    v8f acc[8];
    #pragma unroll
    for (int nt = 0; nt < 8; ++nt) acc[nt] = {};

    for (int k = 0; k < 16; ++k) {          // K = 512 = 16 steps of 32
        // A fragment: row = row0+lh, K = k*32 + hsel*8 + {0..7, 16..23}
        const float* xp = x + (size_t)(row0 + lh) * DIN + k * 32 + hsel * 8;
        f32x4 x0 = *(const f32x4*)(xp + 0);
        f32x4 x1 = *(const f32x4*)(xp + 4);
        f32x4 x2 = *(const f32x4*)(xp + 16);
        f32x4 x3 = *(const f32x4*)(xp + 20);
        v16h a;
        #pragma unroll
        for (int e = 0; e < 4; ++e) {
            a[e]      = (_Float16)x0[e];
            a[4 + e]  = (_Float16)x1[e];
            a[8 + e]  = (_Float16)x2[e];
            a[12 + e] = (_Float16)x3[e];
        }
        #pragma unroll
        for (int nt = 0; nt < 8; ++nt) {
            // B fragment: col = nt*16+lh, K = k*32 + hsel*8 + {0..7, 16..23}
            const _Float16* wp = w1t + (size_t)(nt * 16 + lh) * DIN
                                     + k * 32 + hsel * 8;
            v16h bm = __builtin_shufflevector(*(const h8*)wp, *(const h8*)(wp + 16),
                      0, 1, 2, 3, 4, 5, 6, 7, 8, 9, 10, 11, 12, 13, 14, 15);
            acc[nt] = wmma_f16(a, bm, acc[nt]);
        }
    }

    // -------- epilogue: +b1, ReLU, shared dropout, stash h tile in LDS -----
    #pragma unroll
    for (int nt = 0; nt < 8; ++nt) {
        const int col = nt * 16 + lh;
        const float bias = b1[col];
        #pragma unroll
        for (int r = 0; r < 8; ++r) {
            const int rt = r + 8 * hsel;                  // C layout row
            const uint32_t grow = (uint32_t)(row0 + rt);
            float v = acc[nt][r] + bias;
            v = fmaxf(v, 0.0f);
            uint32_t hs = hash_u32((grow * (uint32_t)H1 + (uint32_t)col) ^ SEED_S);
            v = (hs < TH_KEEP) ? v * INV_KEEP : 0.0f;
            hTile[wid][rt][col] = (_Float16)v;
        }
    }
    __syncthreads();

    // -------- reload h tile as WMMA A fragments (K = 128 = 4 steps) --------
    v16h hA[4];
    #pragma unroll
    for (int kk = 0; kk < 4; ++kk) {
        const _Float16* hp = &hTile[wid][lh][kk * 32 + hsel * 8];
        h8 lo = *(const h8*)(hp);
        h8 hi = *(const h8*)(hp + 16);
        hA[kk] = __builtin_shufflevector(lo, hi,
                 0, 1, 2, 3, 4, 5, 6, 7, 8, 9, 10, 11, 12, 13, 14, 15);
    }

    // -------- aleatoric heads: a_mean = h@Wam+bam ; a_var = exp(h@Wav+bav) -
    #pragma unroll
    for (int nt = 0; nt < 4; ++nt) {
        v8f cm = {}, cv = {};
        #pragma unroll
        for (int kk = 0; kk < 4; ++kk) {
            const size_t off = (size_t)(nt * 16 + lh) * H1 + kk * 32 + hsel * 8;
            v16h bm = __builtin_shufflevector(*(const h8*)(wamt + off), *(const h8*)(wamt + off + 16),
                      0, 1, 2, 3, 4, 5, 6, 7, 8, 9, 10, 11, 12, 13, 14, 15);
            v16h bv = __builtin_shufflevector(*(const h8*)(wavt + off), *(const h8*)(wavt + off + 16),
                      0, 1, 2, 3, 4, 5, 6, 7, 8, 9, 10, 11, 12, 13, 14, 15);
            cm = wmma_f16(hA[kk], bm, cm);
            cv = wmma_f16(hA[kk], bv, cv);
        }
        const int col = nt * 16 + lh;
        const float bm_ = bam[col];
        const float bv_ = bav[col];
        #pragma unroll
        for (int r = 0; r < 8; ++r) {
            const int gr = row0 + r + 8 * hsel;
            out_am[(size_t)gr * H2 + col] = cm[r] + bm_;
            out_av[(size_t)gr * H2 + col] = __expf(cv[r] + bv_);
        }
    }

    // -------- epistemic MC: 10 masked GEMMs, mean + unbiased variance ------
    v16h bE[4][4];                           // keep We^T fragments resident
    #pragma unroll
    for (int nt = 0; nt < 4; ++nt)
        #pragma unroll
        for (int kk = 0; kk < 4; ++kk)
            bE[nt][kk] = __builtin_shufflevector(
                *(const h8*)(wet + (size_t)(nt * 16 + lh) * H1 + kk * 32 + hsel * 8),
                *(const h8*)(wet + (size_t)(nt * 16 + lh) * H1 + kk * 32 + hsel * 8 + 16),
                0, 1, 2, 3, 4, 5, 6, 7, 8, 9, 10, 11, 12, 13, 14, 15);

    v8f sum[4], sq[4];
    #pragma unroll
    for (int nt = 0; nt < 4; ++nt) { sum[nt] = {}; sq[nt] = {}; }

    const uint32_t growA  = (uint32_t)(row0 + lh);   // A-layout row = lane&15
    const _Float16 hscale = (_Float16)INV_KEEP_EFF;
    const _Float16 hzero  = (_Float16)0.0f;

    for (int m = 0; m < MCS; ++m) {
        // one hash -> 4 Bernoulli draws (byte thresholds); K groups of 4
        // per row there are H1/4 = 32 groups; group id = K >> 2
        const uint32_t base32 = ((uint32_t)m * (uint32_t)BDIM + growA) * 32u;
        v16h am[4];
        #pragma unroll
        for (int kk = 0; kk < 4; ++kk) {
            const uint32_t kb  = (uint32_t)(kk * 32 + hsel * 8); // lo-run base
            const uint32_t g0  = base32 + (kb >> 2);             // K kb..kb+3
            uint32_t h0 = hash_u32((g0 + 0) ^ SEED_MC);          // K kb..kb+3
            uint32_t h1 = hash_u32((g0 + 1) ^ SEED_MC);          // K kb+4..kb+7
            uint32_t h2 = hash_u32((g0 + 4) ^ SEED_MC);          // K kb+16..+19
            uint32_t h3 = hash_u32((g0 + 5) ^ SEED_MC);          // K kb+20..+23
            v16h t = hA[kk];
            v16h rv;
            #pragma unroll
            for (int e = 0; e < 16; ++e) {
                uint32_t he = (e < 4) ? h0 : (e < 8) ? h1 : (e < 12) ? h2 : h3;
                uint32_t by = (he >> ((e & 3) * 8)) & 0xFFu;
                rv[e] = (by < TH_EFF8) ? (_Float16)(t[e] * hscale) : hzero;
            }
            am[kk] = rv;
        }
        #pragma unroll
        for (int nt = 0; nt < 4; ++nt) {
            v8f c = {};
            #pragma unroll
            for (int kk = 0; kk < 4; ++kk)
                c = wmma_f16(am[kk], bE[nt][kk], c);
            sum[nt] += c;
            sq[nt]  += c * c;
        }
    }

    #pragma unroll
    for (int nt = 0; nt < 4; ++nt) {
        const int col = nt * 16 + lh;
        const float bias = be[col];
        #pragma unroll
        for (int r = 0; r < 8; ++r) {
            const int gr = row0 + r + 8 * hsel;
            float s   = sum[nt][r];
            float mu  = s * (1.0f / MCS);
            float var = (sq[nt][r] - s * s * (1.0f / MCS)) * (1.0f / (MCS - 1));
            out_em[(size_t)gr * H2 + col] = mu + bias;
            out_ev[(size_t)gr * H2 + col] = var;
        }
    }
}

extern "C" void kernel_launch(void* const* d_in, const int* in_sizes, int n_in,
                              void* d_out, int out_size, void* d_ws, size_t ws_size,
                              hipStream_t stream) {
    const float* x   = (const float*)d_in[0];
    const float* W1  = (const float*)d_in[1];
    const float* b1  = (const float*)d_in[2];
    const float* Wam = (const float*)d_in[3];
    const float* bam = (const float*)d_in[4];
    const float* Wav = (const float*)d_in[5];
    const float* bav = (const float*)d_in[6];
    const float* We  = (const float*)d_in[7];
    const float* be  = (const float*)d_in[8];
    float* out = (float*)d_out;

    // workspace: transposed f16 weights (~180 KB)
    _Float16* w1t  = (_Float16*)d_ws;          // [H1][DIN]
    _Float16* wamt = w1t  + (size_t)DIN * H1;  // [H2][H1]
    _Float16* wavt = wamt + (size_t)H1 * H2;
    _Float16* wet  = wavt + (size_t)H1 * H2;

    uh_xpose_f16<<<(DIN * H1 + 255) / 256, 256, 0, stream>>>(W1,  w1t,  DIN, H1);
    uh_xpose_f16<<<(H1 * H2 + 255) / 256, 256, 0, stream>>>(Wam, wamt, H1, H2);
    uh_xpose_f16<<<(H1 * H2 + 255) / 256, 256, 0, stream>>>(Wav, wavt, H1, H2);
    uh_xpose_f16<<<(H1 * H2 + 255) / 256, 256, 0, stream>>>(We,  wet,  H1, H2);

    // 2048 row-tiles, 8 waves (tiles) per 256-thread block
    uh_fused<<<BDIM / (16 * 8), 256, 0, stream>>>(
        x, b1, bam, bav, be, w1t, wamt, wavt, wet, out);
}
